// TransXL_34703335752252
// MI455X (gfx1250) — matrix operand order repaired and checked by reading, not compiled, not run on hardware
//
#include <hip/hip_runtime.h>
#include <hip/hip_bf16.h>

// ---------------------------------------------------------------------------
// TransfoXL forward for MI455X (gfx1250, wave32, WMMA f32_16x16x32_f16)
// ---------------------------------------------------------------------------

typedef _Float16 half8  __attribute__((ext_vector_type(8)));
typedef _Float16 half16 __attribute__((ext_vector_type(16)));
typedef float    float8 __attribute__((ext_vector_type(8)));

union Frag16 { half16 v; half8 h[2]; };

#define WMMA_F32_F16(A, Bv, C) \
  __builtin_amdgcn_wmma_f32_16x16x32_f16(false, (A), false, (Bv), (short)0, (C), false, false)

// Model constants
constexpr int BATCH_C = 4;
constexpr int QLEN_C  = 512;
constexpr int MLEN_C  = 512;
constexpr int KLEN_C  = 1024;   // MLEN + QLEN
constexpr int DM      = 1024;   // d_model
constexpr int NH      = 16;     // heads
constexpr int DH      = 64;     // d_head
constexpr int NDC     = 1024;   // NH*DH
constexpr int DI      = 4096;   // d_inner
constexpr int QB_ROWS = QLEN_C * BATCH_C;  // 2048
constexpr int KB_ROWS = KLEN_C * BATCH_C;  // 4096

// ---------------------------------------------------------------------------
// Generic f32 -> f16 cast
// ---------------------------------------------------------------------------
__global__ void cast_f16_kernel(const float* __restrict__ src, _Float16* __restrict__ dst, int n) {
  int idx = blockIdx.x * 256 + threadIdx.x;
  if (idx < n) dst[idx] = (_Float16)src[idx];
}

// ---------------------------------------------------------------------------
// Embedding gather: h[(i*B+b)*DM + d] = emb[tok[b][i]*DM + d]
// ---------------------------------------------------------------------------
__global__ void embed_kernel(const int* __restrict__ tok, const float* __restrict__ emb,
                             float* __restrict__ h32, _Float16* __restrict__ h16) {
  int idx = blockIdx.x * 256 + threadIdx.x;      // 0 .. QB_ROWS*DM
  int row = idx >> 10, col = idx & 1023;
  int i = row >> 2, b = row & 3;
  int t = tok[b * QLEN_C + i];
  float v = emb[(size_t)t * DM + col];
  h32[idx] = v;
  h16[idx] = (_Float16)v;
}

// ---------------------------------------------------------------------------
// Positional embedding (f16), pe[k][c]: pos = KLEN-1-k
// ---------------------------------------------------------------------------
__global__ void pe_kernel(_Float16* __restrict__ peh) {
  int idx = blockIdx.x * 256 + threadIdx.x;      // 0 .. KLEN*DM
  int k = idx >> 10, c = idx & 1023;
  float pos = (float)(KLEN_C - 1 - k);
  int ci = c & 511;
  float inv = __powf(10000.0f, -(float)ci * (1.0f / 512.0f));
  float ang = pos * inv;
  float v = (c < 512) ? __sinf(ang) : __cosf(ang);
  peh[idx] = (_Float16)v;
}

// ---------------------------------------------------------------------------
// cat = concat(mems[l], h) as f16 ; rows = kpos*B + b
// ---------------------------------------------------------------------------
__global__ void concat_kernel(const float* __restrict__ memsl, const float* __restrict__ h32,
                              _Float16* __restrict__ cat16) {
  int idx = blockIdx.x * 256 + threadIdx.x;      // 0 .. KB_ROWS*DM
  int row = idx >> 10;
  float v = (row < MLEN_C * BATCH_C) ? memsl[idx]
                                     : h32[idx - MLEN_C * BATCH_C * DM];
  cat16[idx] = (_Float16)v;
}

// ---------------------------------------------------------------------------
// WMMA GEMM: C[M,N] = A[M,K] * W[N,K]^T (f16 in, f32 accumulate)
// Block: 256 threads = 8 waves as 2(M) x 4(N); wave tile 32x32; block tile 64x128.
// Fragment layouts used here (A/B k order contiguous, not the ISA table order):
//   A lane L: row = L&15, halves[0:16)=k0+(L>>4)*16 .. +16 (contiguous)
//   B lane L: col = L&15, halves[0:16)=k0+(L>>4)*16 .. +16 (contiguous)
//   C lane L, vgpr v: row = v + 8*(L>>4), col = L&15
// ---------------------------------------------------------------------------
__global__ void gemm_f16_kernel(const _Float16* __restrict__ A, const _Float16* __restrict__ W,
                                float* __restrict__ outF, _Float16* __restrict__ outH,
                                int M, int N, int K) {
  int lane = threadIdx.x & 31;
  int w    = threadIdx.x >> 5;
  int wm   = w >> 2, wn = w & 3;
  int m0   = blockIdx.y * 64 + wm * 32;
  int n0   = blockIdx.x * 128 + wn * 32;
  int l15  = lane & 15, kg = lane >> 4;

  float8 c[2][2] = {};
  for (int k0 = 0; k0 < K; k0 += 32) {
    Frag16 a[2], b[2];
#pragma unroll
    for (int t = 0; t < 2; ++t) {
      const _Float16* pa = A + (size_t)(m0 + t * 16 + l15) * K + k0;
      a[t].h[0] = *(const half8*)(pa + kg * 16);
      a[t].h[1] = *(const half8*)(pa + kg * 16 + 8);
      const _Float16* pb = W + (size_t)(n0 + t * 16 + l15) * K + k0 + kg * 16;
      b[t].h[0] = *(const half8*)(pb);
      b[t].h[1] = *(const half8*)(pb + 8);
      // WGP-scope prefetch of the next k-tile (locality 3 -> nearest cache level,
      // pulls into all levels per CDNA5 prefetch scope table)
      __builtin_prefetch(pa + 32, 0, 3);
      __builtin_prefetch(pb + 32, 0, 3);
    }
#pragma unroll
    for (int i = 0; i < 2; ++i)
#pragma unroll
      for (int j = 0; j < 2; ++j)
        c[i][j] = WMMA_F32_F16(a[i].v, b[j].v, c[i][j]);
  }
#pragma unroll
  for (int i = 0; i < 2; ++i)
#pragma unroll
    for (int j = 0; j < 2; ++j)
#pragma unroll
      for (int v = 0; v < 8; ++v) {
        int r  = m0 + i * 16 + v + 8 * kg;
        int cc = n0 + j * 16 + l15;
        float val = c[i][j][v];
        if (outF) outF[(size_t)r * N + cc] = val;
        if (outH) outH[(size_t)r * N + cc] = (_Float16)val;
      }
}

// ---------------------------------------------------------------------------
// qb = q + r_w_bias, qr = q + r_r_bias   (q = last QLEN rows of qkv)
// ---------------------------------------------------------------------------
__global__ void qbqr_kernel(const _Float16* __restrict__ qkvh,
                            const float* __restrict__ rwb, const float* __restrict__ rrb,
                            _Float16* __restrict__ qb, _Float16* __restrict__ qr) {
  int idx = blockIdx.x * 256 + threadIdx.x;      // 0 .. QB_ROWS*NDC
  int row = idx >> 10, col = idx & 1023;
  float q = (float)qkvh[(size_t)(row + MLEN_C * BATCH_C) * (3 * NDC) + col];
  qb[idx] = (_Float16)(q + rwb[col]);
  qr[idx] = (_Float16)(q + rrb[col]);
}

// ---------------------------------------------------------------------------
// Fused relative attention, one workgroup per (b, head, 16-query block).
// LDS: score f32[16][1024] | probh f16[16][1024] | vstage f16[256][64]
//      | red f32[256] | outacc f32[16][64]
// ---------------------------------------------------------------------------
__global__ void attn_kernel(const _Float16* __restrict__ qb,    // [QB_ROWS, NDC]
                            const _Float16* __restrict__ qr,    // [QB_ROWS, NDC]
                            const _Float16* __restrict__ qkvh,  // [KB_ROWS, 3*NDC]
                            const _Float16* __restrict__ r16,   // [KLEN, NDC]
                            _Float16* __restrict__ vec16) {     // [QB_ROWS, NDC]
  extern __shared__ char smem[];
  float*    score  = (float*)smem;                               // 64 KB
  _Float16* probh  = (_Float16*)(smem + 65536);                  // 32 KB
  _Float16* vstage = (_Float16*)(smem + 65536 + 32768);          // 32 KB
  float*    red    = (float*)(smem + 65536 + 32768 + 32768);     // 1 KB
  float*    outacc = red + 256;                                  // 4 KB

  int tid  = threadIdx.x;
  int lane = tid & 31, w = tid >> 5;
  int l15  = lane & 15, kg = lane >> 4;

  int blk  = blockIdx.x;                 // B * NH * (QLEN/16) = 2048
  int iblk = blk & 31;
  int n    = (blk >> 5) & 15;
  int b    = blk >> 9;
  int i0   = iblk * 16;

  for (int t = tid; t < 16 * 64; t += 256) outacc[t] = 0.0f;

  // A-fragments (q + biases), constant over the key loop
  Frag16 aqb[2], aqr[2];
#pragma unroll
  for (int t = 0; t < 2; ++t) {
    const _Float16* p1 = qb + (size_t)((i0 + l15) * BATCH_C + b) * NDC + n * DH + t * 32;
    aqb[t].h[0] = *(const half8*)(p1 + kg * 16);
    aqb[t].h[1] = *(const half8*)(p1 + kg * 16 + 8);
    const _Float16* p2 = qr + (size_t)((i0 + l15) * BATCH_C + b) * NDC + n * DH + t * 32;
    aqr[t].h[0] = *(const half8*)(p2 + kg * 16);
    aqr[t].h[1] = *(const half8*)(p2 + kg * 16 + 8);
  }

  // ---- Phase 1: AC = (q + r_w_bias) . k^T -> score LDS ----
  for (int jt = w; jt < 64; jt += 8) {
    float8 c = {};
#pragma unroll
    for (int t = 0; t < 2; ++t) {
      Frag16 bk;
      const _Float16* pk = qkvh + (size_t)((jt * 16 + l15) * BATCH_C + b) * (3 * NDC)
                                + NDC + n * DH + t * 32 + kg * 16;
      bk.h[0] = *(const half8*)(pk);
      bk.h[1] = *(const half8*)(pk + 8);
      c = WMMA_F32_F16(aqb[t].v, bk.v, c);
    }
#pragma unroll
    for (int v = 0; v < 8; ++v)
      score[(v + 8 * kg) * 1024 + jt * 16 + l15] = c[v];
  }
  __syncthreads();

  // ---- Phase 2: BDpre = (q + r_r_bias) . r^T, scatter-add with rel-shift ----
  // BD[i, j] = BDpre[i, j - i + QLEN - 1]  =>  j = jr + i - (QLEN-1)
  for (int jrt = w; jrt < 64; jrt += 8) {
    float8 c = {};
#pragma unroll
    for (int t = 0; t < 2; ++t) {
      Frag16 br;
      const _Float16* pr = r16 + (size_t)(jrt * 16 + l15) * NDC + n * DH + t * 32 + kg * 16;
      br.h[0] = *(const half8*)(pr);
      br.h[1] = *(const half8*)(pr + 8);
      c = WMMA_F32_F16(aqr[t].v, br.v, c);
    }
#pragma unroll
    for (int v = 0; v < 8; ++v) {
      int ri    = v + 8 * kg;
      int i_abs = i0 + ri;
      int j     = (jrt * 16 + l15) + i_abs - (QLEN_C - 1);
      if (j >= 0 && j < KLEN_C) score[ri * 1024 + j] += c[v];  // unique writer per (i,j)
    }
  }
  __syncthreads();

  // ---- Phase 3: scale, mask, softmax (f32), prob -> f16 ----
  {
    int r  = tid >> 4;            // 0..15
    int cc = tid & 15;
    int i_abs = i0 + r;
    int jmax  = i_abs + MLEN_C;
    float m = -3.0e38f;
    for (int k = 0; k < 64; ++k) {
      int j = cc + k * 16;
      float v = score[r * 1024 + j] * 0.125f;    // 1/sqrt(64)
      if (j > jmax) v = -1.0e30f;
      score[r * 1024 + j] = v;
      m = fmaxf(m, v);
    }
    red[r * 16 + cc] = m;
    __syncthreads();
    float rowmax = -3.0e38f;
    for (int k = 0; k < 16; ++k) rowmax = fmaxf(rowmax, red[r * 16 + k]);
    __syncthreads();
    float s = 0.0f;
    for (int k = 0; k < 64; ++k) {
      int j = cc + k * 16;
      float e = __expf(score[r * 1024 + j] - rowmax);
      score[r * 1024 + j] = e;
      s += e;
    }
    red[r * 16 + cc] = s;
    __syncthreads();
    float rowsum = 0.0f;
    for (int k = 0; k < 16; ++k) rowsum += red[r * 16 + k];
    float inv = 1.0f / rowsum;
    for (int k = 0; k < 64; ++k) {
      int j = cc + k * 16;
      probh[r * 1024 + j] = (_Float16)(score[r * 1024 + j] * inv);
    }
  }
  __syncthreads();

  // ---- Phase 4: out = prob . V ; 32 K-chunks of 32, cooperative V staging ----
  float8 cf[4] = {};
  for (int s = 0; s < 4; ++s) {
    {  // stage 8 chunks (256 rows x 64 d) of V into LDS, one row per thread
      int j = (s * 8 + w) * 32 + lane;
      const _Float16* pv = qkvh + (size_t)(j * BATCH_C + b) * (3 * NDC) + 2 * NDC + n * DH;
      half8* dst = (half8*)(vstage + tid * 64);
#pragma unroll
      for (int q8 = 0; q8 < 8; ++q8) dst[q8] = *(const half8*)(pv + q8 * 8);
    }
    __syncthreads();
    Frag16 ap;
    int jcol = (s * 8 + w) * 32;
    const _Float16* pp = probh + (size_t)l15 * 1024 + jcol;
    ap.h[0] = *(const half8*)(pp + kg * 16);
    ap.h[1] = *(const half8*)(pp + kg * 16 + 8);
#pragma unroll
    for (int nt = 0; nt < 4; ++nt) {
      Frag16 bv;
      int d = nt * 16 + l15;
#pragma unroll
      for (int hh = 0; hh < 16; ++hh)
        bv.v[hh] = vstage[(size_t)(w * 32 + kg * 16 + hh) * 64 + d];
      cf[nt] = WMMA_F32_F16(ap.v, bv.v, cf[nt]);
    }
    __syncthreads();
  }
#pragma unroll
  for (int nt = 0; nt < 4; ++nt)
#pragma unroll
    for (int v = 0; v < 8; ++v)
      atomicAdd(&outacc[(v + 8 * kg) * 64 + nt * 16 + l15], cf[nt][v]);
  __syncthreads();
  for (int t = tid; t < 16 * 64; t += 256) {
    int r = t >> 6, d = t & 63;
    vec16[(size_t)((i0 + r) * BATCH_C + b) * NDC + n * DH + d] = (_Float16)outacc[t];
  }
}

// ---------------------------------------------------------------------------
// LayerNorm(x + add [+ bias]) -> out32 (in-place safe), out16
// ---------------------------------------------------------------------------
__global__ void ln_kernel(const float* __restrict__ x, const float* __restrict__ add,
                          const float* __restrict__ bias,
                          const float* __restrict__ g, const float* __restrict__ beta,
                          float* __restrict__ out32, _Float16* __restrict__ out16) {
  __shared__ float sred[256];
  int r = blockIdx.x, tid = threadIdx.x;
  const float* xr = x + (size_t)r * DM;
  const float* ar = add + (size_t)r * DM;

  float s = 0.0f;
  for (int c = tid; c < DM; c += 256)
    s += xr[c] + ar[c] + (bias ? bias[c] : 0.0f);
  sred[tid] = s;
  __syncthreads();
  for (int o = 128; o > 0; o >>= 1) { if (tid < o) sred[tid] += sred[tid + o]; __syncthreads(); }
  float mu = sred[0] * (1.0f / DM);
  __syncthreads();

  float s2 = 0.0f;
  for (int c = tid; c < DM; c += 256) {
    float v = xr[c] + ar[c] + (bias ? bias[c] : 0.0f) - mu;
    s2 += v * v;
  }
  sred[tid] = s2;
  __syncthreads();
  for (int o = 128; o > 0; o >>= 1) { if (tid < o) sred[tid] += sred[tid + o]; __syncthreads(); }
  float rstd = rsqrtf(sred[0] * (1.0f / DM) + 1e-5f);
  __syncthreads();

  for (int c = tid; c < DM; c += 256) {
    float v = (xr[c] + ar[c] + (bias ? bias[c] : 0.0f) - mu) * rstd * g[c] + beta[c];
    out32[(size_t)r * DM + c] = v;
    out16[(size_t)r * DM + c] = (_Float16)v;
  }
}

// ---------------------------------------------------------------------------
// relu(ff1 + b1) -> f16
// ---------------------------------------------------------------------------
__global__ void biasrelu_kernel(const float* __restrict__ gsrc, const float* __restrict__ b1,
                                _Float16* __restrict__ dst) {
  int idx = blockIdx.x * 256 + threadIdx.x;      // 0 .. QB_ROWS*DI
  int col = idx & (DI - 1);
  dst[idx] = (_Float16)fmaxf(gsrc[idx] + b1[col], 0.0f);
}

// ---------------------------------------------------------------------------
// [Q,B,D] (row = i*B+b) -> [B,Q,D] final output
// ---------------------------------------------------------------------------
__global__ void outtr_kernel(const float* __restrict__ h32, float* __restrict__ out) {
  int idx = blockIdx.x * 256 + threadIdx.x;      // 0 .. B*Q*D
  int d = idx & 1023;
  int i = (idx >> 10) & 511;
  int b = idx >> 19;
  out[idx] = h32[((size_t)(i * BATCH_C + b) << 10) + d];
}

// ---------------------------------------------------------------------------
// Host-side orchestration
// ---------------------------------------------------------------------------
static inline size_t align256(size_t x) { return (x + 255) & ~(size_t)255; }

extern "C" void kernel_launch(void* const* d_in, const int* in_sizes, int n_in,
                              void* d_out, int out_size, void* d_ws, size_t ws_size,
                              hipStream_t stream) {
  (void)in_sizes; (void)n_in; (void)out_size; (void)ws_size;

  const int*   tok    = (const int*)d_in[0];
  const float* mems   = (const float*)d_in[1];
  const float* emb    = (const float*)d_in[2];
  const float* qkv_w  = (const float*)d_in[3];
  const float* o_w    = (const float*)d_in[4];
  const float* r_w    = (const float*)d_in[5];
  const float* rwb    = (const float*)d_in[6];
  const float* rrb    = (const float*)d_in[7];
  const float* ln1g   = (const float*)d_in[8];
  const float* ln1b   = (const float*)d_in[9];
  const float* fw1    = (const float*)d_in[10];
  const float* fb1    = (const float*)d_in[11];
  const float* fw2    = (const float*)d_in[12];
  const float* fb2    = (const float*)d_in[13];
  const float* ln2g   = (const float*)d_in[14];
  const float* ln2b   = (const float*)d_in[15];

  char* ws = (char*)d_ws;
  size_t off = 0;
  auto alloc = [&](size_t bytes) { char* p = ws + off; off = align256(off + bytes); return p; };

  _Float16* wqkv16 = (_Float16*)alloc((size_t)3 * NDC * DM * 2);
  _Float16* wo16   = (_Float16*)alloc((size_t)DM * NDC * 2);
  _Float16* wr16   = (_Float16*)alloc((size_t)NDC * DM * 2);
  _Float16* wf1_16 = (_Float16*)alloc((size_t)DI * DM * 2);
  _Float16* wf2_16 = (_Float16*)alloc((size_t)DM * DI * 2);
  float*    h32    = (float*)   alloc((size_t)QB_ROWS * DM * 4);
  _Float16* h16    = (_Float16*)alloc((size_t)QB_ROWS * DM * 2);
  _Float16* cat16  = (_Float16*)alloc((size_t)KB_ROWS * DM * 2);
  _Float16* qkvh   = (_Float16*)alloc((size_t)KB_ROWS * 3 * NDC * 2);
  _Float16* peh    = (_Float16*)alloc((size_t)KLEN_C * DM * 2);
  _Float16* r16    = (_Float16*)alloc((size_t)KLEN_C * NDC * 2);
  _Float16* qb16   = (_Float16*)alloc((size_t)QB_ROWS * NDC * 2);
  _Float16* qr16   = (_Float16*)alloc((size_t)QB_ROWS * NDC * 2);
  _Float16* vec16  = (_Float16*)alloc((size_t)QB_ROWS * NDC * 2);
  _Float16* ffh16  = (_Float16*)alloc((size_t)QB_ROWS * DI * 2);
  float*    g32    = (float*)   alloc((size_t)QB_ROWS * DI * 4);

  const int ATTN_SMEM = 65536 + 32768 + 32768 + 1024 + 4096;  // 136192 B

  embed_kernel<<<(QB_ROWS * DM) / 256, 256, 0, stream>>>(tok, emb, h32, h16);
  pe_kernel<<<(KLEN_C * DM) / 256, 256, 0, stream>>>(peh);

  for (int l = 0; l < 4; ++l) {
    // per-layer weight f16 staging
    cast_f16_kernel<<<(3 * NDC * DM + 255) / 256, 256, 0, stream>>>(qkv_w + (size_t)l * 3 * NDC * DM, wqkv16, 3 * NDC * DM);
    cast_f16_kernel<<<(DM * NDC + 255) / 256, 256, 0, stream>>>(o_w + (size_t)l * DM * NDC, wo16, DM * NDC);
    cast_f16_kernel<<<(NDC * DM + 255) / 256, 256, 0, stream>>>(r_w + (size_t)l * NDC * DM, wr16, NDC * DM);
    cast_f16_kernel<<<(DI * DM + 255) / 256, 256, 0, stream>>>(fw1 + (size_t)l * DI * DM, wf1_16, DI * DM);
    cast_f16_kernel<<<(DM * DI + 255) / 256, 256, 0, stream>>>(fw2 + (size_t)l * DM * DI, wf2_16, DM * DI);

    concat_kernel<<<(KB_ROWS * DM) / 256, 256, 0, stream>>>(mems + (size_t)l * MLEN_C * BATCH_C * DM, h32, cat16);

    // qkv = cat @ qkv_w^T  [4096 x 3072]
    gemm_f16_kernel<<<dim3(3 * NDC / 128, KB_ROWS / 64), 256, 0, stream>>>(cat16, wqkv16, nullptr, qkvh, KB_ROWS, 3 * NDC, DM);
    // r = pe @ r_w^T  [1024 x 1024]
    gemm_f16_kernel<<<dim3(NDC / 128, KLEN_C / 64), 256, 0, stream>>>(peh, wr16, nullptr, r16, KLEN_C, NDC, DM);

    qbqr_kernel<<<(QB_ROWS * NDC) / 256, 256, 0, stream>>>(qkvh, rwb + (size_t)l * NDC, rrb + (size_t)l * NDC, qb16, qr16);

    attn_kernel<<<BATCH_C * NH * (QLEN_C / 16), 256, ATTN_SMEM, stream>>>(qb16, qr16, qkvh, r16, vec16);

    // o-proj: vec @ o_w^T -> g32 ; h = LN(h + g32)
    gemm_f16_kernel<<<dim3(DM / 128, QB_ROWS / 64), 256, 0, stream>>>(vec16, wo16, g32, nullptr, QB_ROWS, DM, NDC);
    ln_kernel<<<QB_ROWS, 256, 0, stream>>>(h32, g32, nullptr, ln1g + (size_t)l * DM, ln1b + (size_t)l * DM, h32, h16);

    // FFN
    gemm_f16_kernel<<<dim3(DI / 128, QB_ROWS / 64), 256, 0, stream>>>(h16, wf1_16, g32, nullptr, QB_ROWS, DI, DM);
    biasrelu_kernel<<<(QB_ROWS * DI) / 256, 256, 0, stream>>>(g32, fb1 + (size_t)l * DI, ffh16);
    gemm_f16_kernel<<<dim3(DM / 128, QB_ROWS / 64), 256, 0, stream>>>(ffh16, wf2_16, g32, nullptr, QB_ROWS, DM, DI);
    ln_kernel<<<QB_ROWS, 256, 0, stream>>>(h32, g32, fb2 + (size_t)l * DM, ln2g + (size_t)l * DM, ln2b + (size_t)l * DM, h32, h16);
  }

  outtr_kernel<<<(BATCH_C * QLEN_C * DM) / 256, 256, 0, stream>>>(h32, (float*)d_out);
}
